// AutoRegressive_45234595562178
// MI455X (gfx1250) — compile-verified, matrix operand order checked by reading, not run on hardware
//
#include <hip/hip_runtime.h>

// ---------------------------------------------------------------------------
// LSTM warmup + autoregressive decode for MI455X (gfx1250, wave32, WMMA).
// 4096 sequences -> 256 waves, 16 sequences per wave (WMMA M-tile).
// Warmup step : 8 N-tiles x (x-path + h-path) = 16 v_wmma_f32_16x16x32_bf16.
//   - bias folded into W_ih row k=13 (x fragment carries 1.0 at k=13)
//   - x A-fragment built in registers from global loads (double-buffered),
//     no LDS staging, accumulators init from inline 0.
// Decode step : x-contribution constant -> 8 register C-tiles (gx) used as
//   accumulator init; 8 gate WMMAs + 1 prediction WMMA per step.
// All weight fragments register-resident (B_ih dies after gx, so warmup and
// decode each stay under 256 VGPRs). Only LDS on the serial chain: the h
// C-layout -> A-layout transpose (16 b16 stores + 2 b128 loads per step).
// ---------------------------------------------------------------------------

typedef __bf16 bf16;
typedef bf16  v16bf __attribute__((ext_vector_type(16)));
typedef float v8f   __attribute__((ext_vector_type(8)));

#define D_IN   13
#define D_H    32
#define NGATES 128          // 4 * D_H
#define TSEQ   512
#define MAXOUT 256
#define BATCH  4096
#define WAVES  2            // waves per block -> 128 blocks across WGPs
#define SEQW   16           // sequences per wave (WMMA M)

__device__ __forceinline__ float fast_sigmoid(float x) {
    return __builtin_amdgcn_rcpf(1.0f + __builtin_amdgcn_exp2f(-1.4426950408889634f * x));
}
__device__ __forceinline__ float fast_tanh(float x) {
    return 2.0f * __builtin_amdgcn_rcpf(1.0f + __builtin_amdgcn_exp2f(-2.8853900817779268f * x)) - 1.0f;
}

__device__ __forceinline__ v8f wmma_bf16(v16bf a, v16bf b, v8f c) {
    return __builtin_amdgcn_wmma_f32_16x16x32_bf16(false, a, false, b, (short)0, c, false, false);
}

// pack two floats into one dword of bf16 pair (v_cvt_pk_bf16_f32)
__device__ __forceinline__ unsigned pk_bf16(float a, float b) {
    union { bf16 h[2]; unsigned u; } v;
    v.h[0] = (bf16)a; v.h[1] = (bf16)b;
    return v.u;
}

// Gather a row-major [rows][32] bf16 tile into the wave32 16-bit A/B fragment
// layout: lane group g holds K-runs [8g,8g+8) and [16+8g,24+8g) -> 2x b128.
struct FragU { union { v16bf v; uint4 q[2]; }; };
__device__ __forceinline__ v16bf load_frag(const bf16* buf, int row, int g) {
    FragU u;
    u.q[0] = *(const uint4*)(buf + row * 32 + 8 * g);
    u.q[1] = *(const uint4*)(buf + row * 32 + 16 + 8 * g);
    return u.v;
}

__global__ __launch_bounds__(WAVES * 32) void lstm_ar_kernel(
    const float* __restrict__ x, const int* __restrict__ lengths,
    const int* __restrict__ out_steps,
    const float* __restrict__ W_ih, const float* __restrict__ W_hh,
    const float* __restrict__ b_ih, const float* __restrict__ b_hh,
    const float* __restrict__ Wd, const float* __restrict__ bd,
    float* __restrict__ out)
{
    __shared__ __align__(16) bf16 sWih[NGATES][32];   // [n][k]; k=13 holds bias
    __shared__ __align__(16) bf16 sWhh[NGATES][32];   // [n][k]
    __shared__ __align__(16) bf16 sWd[16][32];        // [n][k], n>=13 zero
    __shared__ __align__(16) bf16 sX[WAVES][16][32];  // element staging (transition)
    __shared__ __align__(16) bf16 sH[WAVES][16][32];  // per-wave h staging

    const int tid = threadIdx.x;

    // ---- cooperative weight conversion fp32 -> bf16 (once) ----
    for (int i = tid; i < NGATES * 32; i += blockDim.x) {
        int n = i >> 5, k = i & 31;
        float wv = (k < D_IN) ? W_ih[n * D_IN + k]
                 : (k == D_IN) ? (b_ih[n] + b_hh[n])   // bias folded at k=13
                 : 0.0f;
        sWih[n][k] = (bf16)wv;
        sWhh[n][k] = (bf16)W_hh[n * D_H + k];
    }
    for (int i = tid; i < 16 * 32; i += blockDim.x) {
        int n = i >> 5, k = i & 31;
        sWd[n][k] = (bf16)((n < D_IN) ? Wd[n * D_H + k] : 0.0f);
    }
    for (int i = tid; i < WAVES * 16 * 32; i += blockDim.x) {
        (&sX[0][0][0])[i] = (bf16)0.0f;
        (&sH[0][0][0])[i] = (bf16)0.0f;   // h_0 = 0
    }
    __syncthreads();

    const int w    = tid >> 5;
    const int lane = tid & 31;
    const int g    = lane >> 4;
    const int nl   = lane & 15;
    const int s0   = (blockIdx.x * WAVES + w) * SEQW;
    if (s0 >= BATCH) return;

    bf16* myX = &sX[w][0][0];
    bf16* myH = &sH[w][0][0];

    // last valid timestep per sequence (seq = r + 8g for VGPR slot r)
    int lt[8];
#pragma unroll
    for (int r = 0; r < 8; ++r) {
        int L = lengths[s0 + r + 8 * g];
        L = (L < 1) ? 1 : (L > TSEQ ? TSEQ : L);
        lt[r] = L - 1;
    }

    // ---- register-resident weight fragments ----
    v16bf Bih[8], Bhh[8];
#pragma unroll
    for (int t = 0; t < 8; ++t) {
        Bih[t] = load_frag(&sWih[0][0], t * 16 + nl, g);
        Bhh[t] = load_frag(&sWhh[0][0], t * 16 + nl, g);
    }

    // ---- state in WMMA C-tile layout ----
    float h0[8], h1[8], c0[8], c1[8];
#pragma unroll
    for (int r = 0; r < 8; ++r) { h0[r] = h1[r] = c0[r] = c1[r] = 0.0f; }

    // x double-buffer in registers: g=0 lanes carry x[0..7] of row nl,
    // g=1 lanes carry x[8..12] (+ implicit 1.0 at k=13 for the bias fold).
    const int nx = g ? 5 : 8;
    const int xoff = g ? 8 : 0;
    float xr[8];
    {
        const float* xp = x + ((size_t)(s0 + nl) * TSEQ + 0) * D_IN + xoff;
#pragma unroll
        for (int i = 0; i < 8; ++i)
            xr[i] = (i < nx) ? xp[i] : ((g && i == 5) ? 1.0f : 0.0f);
    }

    // ================= warm-up: 512 sequential steps =================
#pragma unroll 1
    for (int t = 0; t < TSEQ; ++t) {
        // build x A-fragment in registers (no LDS)
        FragU ux;
        ux.q[0] = make_uint4(pk_bf16(xr[0], xr[1]), pk_bf16(xr[2], xr[3]),
                             pk_bf16(xr[4], xr[5]), pk_bf16(xr[6], xr[7]));
        ux.q[1] = make_uint4(0u, 0u, 0u, 0u);      // K 16..23 / 24..31 are zero
        const v16bf Ax = ux.v;

        // issue next step's global loads early (latency off the chain)
        {
            int tn = (t + 1 < TSEQ) ? t + 1 : t;
            const float* xp = x + ((size_t)(s0 + nl) * TSEQ + tn) * D_IN + xoff;
#pragma unroll
            for (int i = 0; i < 8; ++i) if (i < nx) xr[i] = xp[i];
        }

        const v16bf Ah = load_frag(myH, nl, g);

#pragma unroll
        for (int half = 0; half < 2; ++half) {
            v8f z = {};
            v8f ai = wmma_bf16(Ax, Bih[0 + half], z);
            v8f af = wmma_bf16(Ax, Bih[2 + half], z);
            v8f ag = wmma_bf16(Ax, Bih[4 + half], z);
            v8f ao = wmma_bf16(Ax, Bih[6 + half], z);
            ai = wmma_bf16(Ah, Bhh[0 + half], ai);
            af = wmma_bf16(Ah, Bhh[2 + half], af);
            ag = wmma_bf16(Ah, Bhh[4 + half], ag);
            ao = wmma_bf16(Ah, Bhh[6 + half], ao);

            float* cc = half ? c1 : c0;
            float* hh = half ? h1 : h0;
#pragma unroll
            for (int r = 0; r < 8; ++r) {
                float iv = fast_sigmoid(ai[r]);
                float fv = fast_sigmoid(af[r]);
                float gv = fast_tanh(ag[r]);
                float ov = fast_sigmoid(ao[r]);
                float cn = fv * cc[r] + iv * gv;
                float hn = ov * fast_tanh(cn);
                bool keep = (t <= lt[r]);          // freeze past last valid step
                cc[r] = keep ? cn : cc[r];
                hh[r] = keep ? hn : hh[r];
                myH[(r + 8 * g) * 32 + nl + 16 * half] = (bf16)hh[r];
            }
        }
    }

    // ---- decode-only metadata (kept out of warmup's register budget) ----
    int ost[8];
#pragma unroll
    for (int r = 0; r < 8; ++r) {
        int S = out_steps[s0 + r + 8 * g];
        ost[r] = (S < 1) ? 1 : (S > MAXOUT ? MAXOUT : S);
    }
    const float biasd = (nl < D_IN) ? bd[nl] : 0.0f;
    const v16bf Bd = load_frag(&sWd[0][0], nl, g);

    // ================= element = h_sel @ Wd^T + bd =================
    v16bf Ah = load_frag(myH, nl, g);
    v8f elem;
#pragma unroll
    for (int r = 0; r < 8; ++r) elem[r] = biasd;
    elem = wmma_bf16(Ah, Bd, elem);

    // step-0 output + stage element (constant decode input, 1.0 at k=13)
#pragma unroll
    for (int r = 0; r < 8; ++r) {
        int m = r + 8 * g;
        myX[m * 32 + nl] = (bf16)((nl == D_IN) ? 1.0f : elem[r]);
        if (nl < D_IN)
            out[((size_t)(s0 + m) * MAXOUT + 0) * D_IN + nl] = elem[r];
    }
    const v16bf Ae = load_frag(myX, nl, g);

    // gx[t] = elem @ W_ih^T + b (bias via k=13 fold): register C-tiles
    v8f gx[8];
#pragma unroll
    for (int t = 0; t < 8; ++t) {
        v8f z = {};
        gx[t] = wmma_bf16(Ae, Bih[t], z);          // last use of Bih
    }

    // ================= decode: 255 sequential steps =================
#pragma unroll 1
    for (int s = 1; s < MAXOUT; ++s) {
#pragma unroll
        for (int half = 0; half < 2; ++half) {
            v8f ai = wmma_bf16(Ah, Bhh[0 + half], gx[0 + half]);
            v8f af = wmma_bf16(Ah, Bhh[2 + half], gx[2 + half]);
            v8f ag = wmma_bf16(Ah, Bhh[4 + half], gx[4 + half]);
            v8f ao = wmma_bf16(Ah, Bhh[6 + half], gx[6 + half]);

            float* cc = half ? c1 : c0;
            float* hh = half ? h1 : h0;
#pragma unroll
            for (int r = 0; r < 8; ++r) {
                float iv = fast_sigmoid(ai[r]);
                float fv = fast_sigmoid(af[r]);
                float gv = fast_tanh(ag[r]);
                float ov = fast_sigmoid(ao[r]);
                float cn = fv * cc[r] + iv * gv;
                float hn = ov * fast_tanh(cn);
                cc[r] = cn;
                hh[r] = hn;
                myH[(r + 8 * g) * 32 + nl + 16 * half] = (bf16)hn;
            }
        }
        Ah = load_frag(myH, nl, g);                // new h as A-fragment

        v8f pr;
#pragma unroll
        for (int r = 0; r < 8; ++r) pr[r] = biasd;
        pr = wmma_bf16(Ah, Bd, pr);                // pred = h @ Wd^T + bd

#pragma unroll
        for (int r = 0; r < 8; ++r) {
            int m = r + 8 * g;
            if (nl < D_IN)
                out[((size_t)(s0 + m) * MAXOUT + s) * D_IN + nl] =
                    (s < ost[r]) ? pr[r] : 0.0f;
        }
    }
}

extern "C" void kernel_launch(void* const* d_in, const int* in_sizes, int n_in,
                              void* d_out, int out_size, void* d_ws, size_t ws_size,
                              hipStream_t stream) {
    // setup_inputs order: x, lengths, out_steps, max_out, W_ih, W_hh, b_ih, b_hh, Wd, bd
    const float* x        = (const float*)d_in[0];
    const int*   lengths  = (const int*)d_in[1];
    const int*   outsteps = (const int*)d_in[2];
    // d_in[3] = max_out (scalar, constant 256)
    const float* W_ih     = (const float*)d_in[4];
    const float* W_hh     = (const float*)d_in[5];
    const float* b_ih     = (const float*)d_in[6];
    const float* b_hh     = (const float*)d_in[7];
    const float* Wd       = (const float*)d_in[8];
    const float* bd       = (const float*)d_in[9];
    float*       out      = (float*)d_out;

    const int blocks = BATCH / (WAVES * SEQW);   // 4096 / 32 = 128
    hipLaunchKernelGGL(lstm_ar_kernel, dim3(blocks), dim3(WAVES * 32), 0, stream,
                       x, lengths, outsteps, W_ih, W_hh, b_ih, b_hh, Wd, bd, out);
}
